// KernelEncoderLayer_78460462563867
// MI455X (gfx1250) — matrix operand order repaired and checked by reading, not compiled, not run on hardware
//
#include <hip/hip_runtime.h>

// ---------------------------------------------------------------------------
// Fused point-cloud kernel-conv encoder layer for gfx1250 (MI455X).
// All large GEMMs on v_wmma_f32_16x16x32_f16 (f16 in / f32 acc), kmat built
// on the fly with v_exp_f32 directly into WMMA A-fragments, comp_w staged in
// workspace pre-packed in the WMMA B-fragment register layout.
// K2 splits each tile's 288-chunk contraction across 4 waves (1024 waves
// total -> fills every SIMD32) with a deterministic LDS partial reduction,
// and stages positions into LDS with GLOBAL_LOAD_ASYNC_TO_LDS_B128
// (ASYNCcnt-tracked LDS DMA; falls back to synchronous staging if absent).
// ---------------------------------------------------------------------------

#define DEV __device__ __forceinline__

typedef __attribute__((ext_vector_type(16))) _Float16 v16h;
typedef __attribute__((ext_vector_type(8)))  float    v8f;

#if __has_builtin(__builtin_amdgcn_global_load_async_to_lds_b128)
#define HAVE_ASYNC_LDS 1
// exact pointee type per clang's declared prototype: int __vector_size__(16)
typedef __attribute__((__vector_size__(16))) int i32x4;
typedef __attribute__((address_space(1))) i32x4 i32x4_g;  // global src
typedef __attribute__((address_space(3))) i32x4 i32x4_l;  // LDS dst
#else
#define HAVE_ASYNC_LDS 0
#endif

constexpr int Bb   = 4;
constexpr int Nn   = 1024;
constexpr int Kk   = 9;
constexpr int Cc   = 32;
constexpr int CMd  = 128;
constexpr int ROWS = Bb * Nn;       // 4096 points
constexpr float KEXP = -0.72134752044448169f;  // -0.5/sigma^2 * log2(e), sigma=1

// workspace layout (bytes, all 128B-aligned)
constexpr size_t OFF_WSB  = 0;        // comp_w packed B-frags: 2304 frags * 32 lanes * 32B = 2359296
constexpr size_t OFF_YPRE = 2359296;  // y after leaky, pre-BN: 4096*32*4 = 524288
constexpr size_t OFF_M32  = 2883584;  // mean[32]
constexpr size_t OFF_R32  = 2883712;  // rstd[32]
constexpr size_t OFF_YBN  = 2883840;  // y after BN + residual: 524288
constexpr size_t OFF_WSA  = 3408128;  // y_bn f16 A-frags: 256 tiles * 32 lanes * 32B = 262144
constexpr size_t OFF_W1F  = 3670272;  // w1 B-frags: 8*32*32B = 8192
constexpr size_t OFF_W2F  = 3678464;  // w2 B-frags: 8*32*32B = 8192
constexpr size_t OFF_H    = 3686656;  // hidden: 4096*128*4 = 2097152
constexpr size_t OFF_S1   = 5783808;  // bn1 scale[128]
constexpr size_t OFF_T1   = 5784320;  // bn1 shift[128]

// 16-bit A 16x32 k order chosen here: K cols sequential per lane-half (matches kmapB)
DEV int kmapA(int e, int H) {
  return e + (H << 4);
}
// 16-bit B 32x16 k order chosen here: K rows sequential per lane-half
DEV int kmapB(int e, int H) { return e + (H << 4); }

DEV float lrelu(float v) { return v > 0.0f ? v : 0.01f * v; }

DEV float fexp2(float x) {
#if __has_builtin(__builtin_amdgcn_exp2f)
  return __builtin_amdgcn_exp2f(x);
#else
  return exp2f(x);
#endif
}

DEV v8f wmma16(v16h a, v16h b, v8f c) {
  return __builtin_amdgcn_wmma_f32_16x16x32_f16(false, a, false, b, (short)0, c,
                                                false, false);
}

// ---------------------------------------------------------------------------
// K0: pre-pack w1 [32x128] and w2 [128x32] into f16 B-fragment layout.
// 1 block x 512 threads = 16 waves, one fragment per wave.
// ---------------------------------------------------------------------------
__global__ void k0_prep_w(const float* __restrict__ w1, const float* __restrict__ w2,
                          v16h* __restrict__ w1f, v16h* __restrict__ w2f) {
  int lane = threadIdx.x & 31, wv = threadIdx.x >> 5;
  int c16 = lane & 15, H = lane >> 4;
  v16h f;
  if (wv < 8) {                      // w1 fragment wv: cols wv*16..wv*16+15
#pragma unroll
    for (int e = 0; e < 16; ++e)
      f[e] = (_Float16)w1[kmapB(e, H) * CMd + wv * 16 + c16];
    w1f[wv * 32 + lane] = f;
  } else {                           // w2 fragment: chunk q (ci group), col half ch
    int q = (wv - 8) >> 1, ch = (wv - 8) & 1;
#pragma unroll
    for (int e = 0; e < 16; ++e)
      f[e] = (_Float16)w2[(q * 32 + kmapB(e, H)) * Cc + ch * 16 + c16];
    w2f[(q * 2 + ch) * 32 + lane] = f;
  }
}

// ---------------------------------------------------------------------------
// K1: comp_w producer. comp_w_k = W[b] (NxC) @ conv_w[k] (CxC) via WMMA,
// LDS bounce 32x32 f16 tile, store in consumer B-fragment layout.
// grid 288 x 128 threads (4 waves/block), wave = (b,k,n-block of 32).
// ---------------------------------------------------------------------------
__global__ void k1_compw(const float* __restrict__ wts, const float* __restrict__ cw,
                         v16h* __restrict__ wsB) {
  __shared__ _Float16 tl[4][32][33];
  int lane = threadIdx.x & 31, wv = threadIdx.x >> 5;
  int c16 = lane & 15, H = lane >> 4;
  int g = blockIdx.x * 4 + wv;        // 0..1151
  int b = g / 288, rem = g % 288;
  int k = rem >> 5, nbi = rem & 31;
  int nb = nbi * 32;

  v16h a0, a1, bc0, bc1;
#pragma unroll
  for (int e = 0; e < 16; ++e) {
    int ci = kmapA(e, H);
    a0[e] = (_Float16)wts[(b * Nn + nb + c16) * Cc + ci];
    a1[e] = (_Float16)wts[(b * Nn + nb + 16 + c16) * Cc + ci];
    int cj = kmapB(e, H);
    bc0[e] = (_Float16)cw[(k * Cc + cj) * Cc + c16];
    bc1[e] = (_Float16)cw[(k * Cc + cj) * Cc + 16 + c16];
  }
  v8f acc00 = {}, acc01 = {}, acc10 = {}, acc11 = {};
  acc00 = wmma16(a0, bc0, acc00);
  acc01 = wmma16(a0, bc1, acc01);
  acc10 = wmma16(a1, bc0, acc10);
  acc11 = wmma16(a1, bc1, acc11);

  // D-layout -> LDS tile [n_local][c]
#pragma unroll
  for (int r = 0; r < 8; ++r) {
    tl[wv][8 * H + r][c16]           = (_Float16)acc00[r];
    tl[wv][8 * H + r][16 + c16]      = (_Float16)acc01[r];
    tl[wv][16 + 8 * H + r][c16]      = (_Float16)acc10[r];
    tl[wv][16 + 8 * H + r][16 + c16] = (_Float16)acc11[r];
  }
  __syncthreads();

  // re-read in B-fragment layout, store packed (32B/lane, fully coalesced)
  v16h o0, o1;
#pragma unroll
  for (int e = 0; e < 16; ++e) {
    int nl = e + 16 * H;
    o0[e] = tl[wv][nl][c16];
    o1[e] = tl[wv][nl][16 + c16];
  }
  size_t fi = ((size_t)(b * Kk + k) * 32 + nbi) * 2;
  wsB[fi * 32 + lane]       = o0;
  wsB[(fi + 1) * 32 + lane] = o1;
}

// ---------------------------------------------------------------------------
// K2: main contraction. Each 16-row i-tile is computed by 4 waves, each
// covering 72 of the 288 (k, n-block) chunks; per chunk a wave builds the
// kmat A-fragment with v_exp_f32 from LDS-staged positions and issues 2
// WMMAs. Partials are combined deterministically through LDS.
// grid 128 x 256 (8 waves = 2 tiles per block) -> 1024 waves, full machine.
// ---------------------------------------------------------------------------
__global__ void k2_main(const float* __restrict__ pos, const float* __restrict__ kpos,
                        const v16h* __restrict__ wsB, float* __restrict__ ypre) {
  __shared__ float2 spos[Nn];             // 8 KB: positions of this batch
  __shared__ float part[2][4][16][32];    // 16 KB: per-wave partial tiles
  int lane = threadIdx.x & 31, wv = threadIdx.x >> 5;
  int c16 = lane & 15, H = lane >> 4;
  int tsel = wv >> 2;                      // tile within block (0,1)
  int wsub = wv & 3;                       // contraction splitter (0..3)
  int tileg = blockIdx.x * 2 + tsel;       // global tile 0..255
  int b = (blockIdx.x * 2) >> 6;           // 64 tiles per batch; both tiles same b

  const float2* gp = (const float2*)pos + b * Nn;
#if HAVE_ASYNC_LDS
  {
    // 8 KB staged via ASYNCcnt-tracked LDS DMA: 256 lanes x 2 x b128
    char* gbase = (char*)gp;
    char* lbase = (char*)spos;
    __builtin_amdgcn_global_load_async_to_lds_b128(
        (i32x4_g*)(gbase + threadIdx.x * 16),
        (i32x4_l*)(lbase + threadIdx.x * 16), 0, 0);
    __builtin_amdgcn_global_load_async_to_lds_b128(
        (i32x4_g*)(gbase + 4096 + threadIdx.x * 16),
        (i32x4_l*)(lbase + 4096 + threadIdx.x * 16), 0, 0);
#if __has_builtin(__builtin_amdgcn_s_wait_asynccnt)
    __builtin_amdgcn_s_wait_asynccnt(0);
#else
    asm volatile("s_wait_asynccnt 0x0" ::: "memory");
#endif
  }
#else
  for (int i = threadIdx.x; i < Nn; i += 256) spos[i] = gp[i];
#endif
  __syncthreads();

  float2 pi = spos[(tileg & 63) * 16 + c16];
  const float2* kp = (const float2*)kpos;
  v8f acc0 = {}, acc1 = {};

  size_t fb0 = (size_t)b * Kk * 64;
  for (int ch = wsub * 72; ch < wsub * 72 + 72; ++ch) {
    int k = ch >> 5, nbi = ch & 31;
    float kx = kp[k].x, ky = kp[k].y;
    int nb = nbi << 5;
    v16h a;
#pragma unroll
    for (int p = 0; p < 8; ++p) {
      int n = nb + kmapA(2 * p, H);            // even base -> pair n, n+1
      float2 q0 = spos[n], q1 = spos[n + 1];
      float dx0 = pi.x - q0.x - kx, dy0 = pi.y - q0.y - ky;
      float dx1 = pi.x - q1.x - kx, dy1 = pi.y - q1.y - ky;
      a[2 * p]     = (_Float16)fexp2((dx0 * dx0 + dy0 * dy0) * KEXP);
      a[2 * p + 1] = (_Float16)fexp2((dx1 * dx1 + dy1 * dy1) * KEXP);
    }
    size_t fi = fb0 + (size_t)ch * 2;
    v16h b0 = wsB[fi * 32 + lane];
    v16h b1 = wsB[(fi + 1) * 32 + lane];
    __builtin_prefetch(&wsB[(fi + 2) * 32 + lane], 0, 1);  // global_prefetch
    acc0 = wmma16(a, b0, acc0);
    acc1 = wmma16(a, b1, acc1);
  }

  // deposit this wave's partial tile (D-layout -> [row][col])
#pragma unroll
  for (int r = 0; r < 8; ++r) {
    part[tsel][wsub][8 * H + r][c16]      = acc0[r];
    part[tsel][wsub][8 * H + r][16 + c16] = acc1[r];
  }
  __syncthreads();

  // deterministic fixed-order combine + leaky ReLU + store
  for (int i = threadIdx.x; i < 1024; i += 256) {
    int ts = i >> 9, rem = i & 511, rr = rem >> 5, cc = rem & 31;
    float v = ((part[ts][0][rr][cc] + part[ts][1][rr][cc]) +
               (part[ts][2][rr][cc] + part[ts][3][rr][cc]));
    int row = (blockIdx.x * 2 + ts) * 16 + rr;
    ypre[row * Cc + cc] = lrelu(v);
  }
}

// ---------------------------------------------------------------------------
// K3: BN stats over 4096 points, 32 channels (block = channel).
// ---------------------------------------------------------------------------
__global__ void k3_stats32(const float* __restrict__ ypre, float* __restrict__ m32,
                           float* __restrict__ r32) {
  __shared__ float ssum[256], ssq[256];
  int c = blockIdx.x;
  float s = 0.0f, q = 0.0f;
  for (int r = threadIdx.x; r < ROWS; r += 256) {
    float v = ypre[r * Cc + c];
    s += v; q += v * v;
  }
  ssum[threadIdx.x] = s; ssq[threadIdx.x] = q;
  __syncthreads();
  for (int off = 128; off > 0; off >>= 1) {
    if (threadIdx.x < off) {
      ssum[threadIdx.x] += ssum[threadIdx.x + off];
      ssq[threadIdx.x]  += ssq[threadIdx.x + off];
    }
    __syncthreads();
  }
  if (threadIdx.x == 0) {
    float mean = ssum[0] / (float)ROWS;
    float var  = ssq[0] / (float)ROWS - mean * mean;
    m32[c] = mean;
    r32[c] = rsqrtf(var + 1e-5f);
  }
}

// ---------------------------------------------------------------------------
// K4: y = BN(y)+weights; write f32 copy and f16 A-fragment copy for the MLP.
// thread = one point row. grid 16 x 256.
// ---------------------------------------------------------------------------
__global__ void k4_bn_res(const float* __restrict__ ypre, const float* __restrict__ wts,
                          const float* __restrict__ m32, const float* __restrict__ r32,
                          const float* __restrict__ gam, const float* __restrict__ bet,
                          float* __restrict__ ybn, v16h* __restrict__ wsA) {
  __shared__ float sg[Cc], sb[Cc], sm[Cc], sr[Cc];
  if (threadIdx.x < Cc) {
    sg[threadIdx.x] = gam[threadIdx.x];
    sb[threadIdx.x] = bet[threadIdx.x];
    sm[threadIdx.x] = m32[threadIdx.x];
    sr[threadIdx.x] = r32[threadIdx.x];
  }
  __syncthreads();
  int row = blockIdx.x * 256 + threadIdx.x;
  float y[Cc];
#pragma unroll
  for (int c = 0; c < Cc; ++c) {
    float v = ypre[row * Cc + c];
    v = sg[c] * (v - sm[c]) * sr[c] + sb[c] + wts[row * Cc + c];
    y[c] = v;
    ybn[row * Cc + c] = v;
  }
  int tile = row >> 4, l = row & 15;
  v16h a0, a1;
#pragma unroll
  for (int e = 0; e < 16; ++e) {
    a0[e] = (_Float16)y[kmapA(e, 0)];
    a1[e] = (_Float16)y[kmapA(e, 1)];
  }
  wsA[tile * 32 + l]      = a0;
  wsA[tile * 32 + 16 + l] = a1;
}

// ---------------------------------------------------------------------------
// K5: h = leaky(y_bn @ w1 + b1). M=4096 K=32 N=128; wave = 16-row tile, 8 WMMAs.
// ---------------------------------------------------------------------------
__global__ void k5_mlp1(const v16h* __restrict__ wsA, const v16h* __restrict__ w1f,
                        const float* __restrict__ b1, float* __restrict__ h) {
  int lane = threadIdx.x & 31, wv = threadIdx.x >> 5;
  int c16 = lane & 15, H = lane >> 4;
  int tile = blockIdx.x * 8 + wv;
  v16h a = wsA[tile * 32 + lane];
  int gr0 = tile * 16 + 8 * H;
#pragma unroll
  for (int f = 0; f < 8; ++f) {
    v8f acc = {};
    acc = wmma16(a, w1f[f * 32 + lane], acc);
    int c = f * 16 + c16;
    float bias = b1[c];
#pragma unroll
    for (int r = 0; r < 8; ++r)
      h[(gr0 + r) * CMd + c] = lrelu(acc[r] + bias);
  }
}

// ---------------------------------------------------------------------------
// K6: bn1 stats over 4096 rows, 128 channels; emits affine s[c], t[c].
// ---------------------------------------------------------------------------
__global__ void k6_stats128(const float* __restrict__ h, const float* __restrict__ g1,
                            const float* __restrict__ be1, float* __restrict__ s1,
                            float* __restrict__ t1) {
  __shared__ float ssum[256], ssq[256];
  int c = blockIdx.x;
  float s = 0.0f, q = 0.0f;
  for (int r = threadIdx.x; r < ROWS; r += 256) {
    float v = h[r * CMd + c];
    s += v; q += v * v;
  }
  ssum[threadIdx.x] = s; ssq[threadIdx.x] = q;
  __syncthreads();
  for (int off = 128; off > 0; off >>= 1) {
    if (threadIdx.x < off) {
      ssum[threadIdx.x] += ssum[threadIdx.x + off];
      ssq[threadIdx.x]  += ssq[threadIdx.x + off];
    }
    __syncthreads();
  }
  if (threadIdx.x == 0) {
    float mean = ssum[0] / (float)ROWS;
    float var  = ssq[0] / (float)ROWS - mean * mean;
    float sc = g1[c] * rsqrtf(var + 1e-5f);
    s1[c] = sc;
    t1[c] = be1[c] - mean * sc;
  }
}

// ---------------------------------------------------------------------------
// K7: out = y_bn + (s*h + t) @ w2 + b2. bn1 affine folded into A-fragment.
// M=4096 K=128 N=32; wave = 16-row tile, 4 chunks x 2 WMMAs.
// ---------------------------------------------------------------------------
__global__ void k7_final(const float* __restrict__ h, const v16h* __restrict__ w2f,
                         const float* __restrict__ s1, const float* __restrict__ t1,
                         const float* __restrict__ ybn, const float* __restrict__ b2,
                         float* __restrict__ out) {
  __shared__ float ss[CMd], st[CMd];
  if (threadIdx.x < CMd) {
    ss[threadIdx.x] = s1[threadIdx.x];
    st[threadIdx.x] = t1[threadIdx.x];
  }
  __syncthreads();
  int lane = threadIdx.x & 31, wv = threadIdx.x >> 5;
  int c16 = lane & 15, H = lane >> 4;
  int tile = blockIdx.x * 8 + wv;
  int row = tile * 16 + c16;
  v8f acc0 = {}, acc1 = {};
#pragma unroll
  for (int q = 0; q < 4; ++q) {
    v16h a;
#pragma unroll
    for (int e = 0; e < 16; ++e) {
      int ci = q * 32 + kmapA(e, H);
      float hv = h[row * CMd + ci];
      a[e] = (_Float16)(ss[ci] * hv + st[ci]);
    }
    acc0 = wmma16(a, w2f[(q * 2 + 0) * 32 + lane], acc0);
    acc1 = wmma16(a, w2f[(q * 2 + 1) * 32 + lane], acc1);
  }
  float bb0 = b2[c16], bb1 = b2[c16 + 16];
  int gr0 = tile * 16 + 8 * H;
#pragma unroll
  for (int r = 0; r < 8; ++r) {
    int rr = gr0 + r;
    out[rr * Cc + c16]      = ybn[rr * Cc + c16]      + acc0[r] + bb0;
    out[rr * Cc + c16 + 16] = ybn[rr * Cc + c16 + 16] + acc1[r] + bb1;
  }
}

// ---------------------------------------------------------------------------
extern "C" void kernel_launch(void* const* d_in, const int* in_sizes, int n_in,
                              void* d_out, int out_size, void* d_ws, size_t ws_size,
                              hipStream_t stream) {
  (void)in_sizes; (void)n_in; (void)out_size; (void)ws_size;
  const float* positions  = (const float*)d_in[0];
  const float* weights    = (const float*)d_in[1];
  const float* kernel_pos = (const float*)d_in[2];
  const float* conv_w     = (const float*)d_in[3];
  const float* bn_gamma   = (const float*)d_in[4];
  const float* bn_beta    = (const float*)d_in[5];
  const float* w1         = (const float*)d_in[6];
  const float* b1         = (const float*)d_in[7];
  const float* bn1_gamma  = (const float*)d_in[8];
  const float* bn1_beta   = (const float*)d_in[9];
  const float* w2         = (const float*)d_in[10];
  const float* b2         = (const float*)d_in[11];

  char* ws = (char*)d_ws;
  v16h*  wsB  = (v16h*)(ws + OFF_WSB);
  float* ypre = (float*)(ws + OFF_YPRE);
  float* m32  = (float*)(ws + OFF_M32);
  float* r32  = (float*)(ws + OFF_R32);
  float* ybn  = (float*)(ws + OFF_YBN);
  v16h*  wsA  = (v16h*)(ws + OFF_WSA);
  v16h*  w1f  = (v16h*)(ws + OFF_W1F);
  v16h*  w2f  = (v16h*)(ws + OFF_W2F);
  float* hbuf = (float*)(ws + OFF_H);
  float* s1   = (float*)(ws + OFF_S1);
  float* t1   = (float*)(ws + OFF_T1);
  float* out  = (float*)d_out;

  k0_prep_w  <<<1,   512, 0, stream>>>(w1, w2, w1f, w2f);
  k1_compw   <<<288, 128, 0, stream>>>(weights, conv_w, wsB);
  k2_main    <<<128, 256, 0, stream>>>(positions, kernel_pos, wsB, ypre);
  k3_stats32 <<<32,  256, 0, stream>>>(ypre, m32, r32);
  k4_bn_res  <<<16,  256, 0, stream>>>(ypre, weights, m32, r32, bn_gamma, bn_beta, ybn, wsA);
  k5_mlp1    <<<32,  256, 0, stream>>>(wsA, w1f, b1, hbuf);
  k6_stats128<<<128, 256, 0, stream>>>(hbuf, bn1_gamma, bn1_beta, s1, t1);
  k7_final   <<<32,  256, 0, stream>>>(hbuf, w2f, s1, t1, ybn, b2, out);
}
